// SMPLModel_76776835384020
// MI455X (gfx1250) — compile-verified, matrix operand order checked by reading, not run on hardware
//
#include <hip/hip_runtime.h>

#define B_ 512
#define V_ 6890
#define NB_ 10
#define NJ_ 24
#define NEX_ 9
#define NOUT_ (V_ + NJ_ + NEX_)   // 6923 rows per batch in output

#define TILES_ 4                   // 16-vertex WMMA tiles per wave
#define VPB_ (8 * TILES_ * 16)     // vertices per 256-thread block = 512

typedef _Float16 v16h __attribute__((ext_vector_type(16)));
typedef float    v8f  __attribute__((ext_vector_type(8)));

__constant__ int c_par[NJ_] = {-1, 0, 0, 0, 1, 2, 3, 4, 5, 6, 7, 8, 9, 9, 9,
                               12, 13, 14, 16, 17, 18, 19, 20, 21};

// ---------------------------------------------------------------------------
// K1: precompute Jt = J_reg @ v_template(s)  and  JS = J_reg @ shapedirs.
// 72 blocks, one per (joint j, coord k); each block reduces over V.
// JtJS layout: [0..71] = Jt[j*3+k]; [72 + (j*3+k)*10 + l] = JS.
// ---------------------------------------------------------------------------
__global__ void k1_jreg(const float* __restrict__ scale,
                        const float* __restrict__ smpl,
                        const float* __restrict__ smil,
                        const float* __restrict__ SD,
                        const float* __restrict__ Jreg,
                        float* __restrict__ JtJS) {
    int bx = blockIdx.x;              // 0..71
    int j = bx / 3, k = bx % 3;
    int tid = threadIdx.x;            // 256 threads
    float s = scale[0];
    float acc[11];
#pragma unroll
    for (int c = 0; c < 11; ++c) acc[c] = 0.f;
    for (int v = tid; v < V_; v += 256) {
        float r  = Jreg[(size_t)j * V_ + v];
        float vt = smpl[(size_t)v * 3 + k] * s + (1.f - s) * smil[(size_t)v * 3 + k];
        acc[0] += r * vt;
        const float* sd = SD + ((size_t)v * 3 + k) * NB_;
#pragma unroll
        for (int l = 0; l < NB_; ++l) acc[1 + l] += r * sd[l];
    }
    __shared__ float red[256 * 11];
#pragma unroll
    for (int c = 0; c < 11; ++c) red[tid * 11 + c] = acc[c];
    __syncthreads();
    for (int off = 128; off > 0; off >>= 1) {
        if (tid < off) {
#pragma unroll
            for (int c = 0; c < 11; ++c)
                red[tid * 11 + c] += red[(tid + off) * 11 + c];
        }
        __syncthreads();
    }
    if (tid == 0) {
        JtJS[bx] = red[0];
        for (int l = 0; l < NB_; ++l) JtJS[72 + bx * NB_ + l] = red[1 + l];
    }
}

// ---------------------------------------------------------------------------
// K3: per-batch J_shaped, Rodrigues, kinematic chain, A matrices (3x4 each),
// and posed joints (+transl) written straight to the output.
// One block (128 threads) per batch.
// ---------------------------------------------------------------------------
__global__ void k3_chain(const float* __restrict__ betas,
                         const float* __restrict__ body_pose,
                         const float* __restrict__ gorient,
                         const float* __restrict__ transl,
                         const float* __restrict__ JtJS,
                         float* __restrict__ A_ws,
                         float* __restrict__ out) {
    int b = blockIdx.x;
    int tid = threadIdx.x;
    __shared__ float Js[NJ_ * 3];
    __shared__ float Rm[NJ_ * 9];
    __shared__ float Ch[NJ_ * 12];

    if (tid < 72) {                      // J_shaped = (Jt + JS·betas[1:]) * betas[0]
        float a = JtJS[tid];
#pragma unroll
        for (int l = 0; l < NB_; ++l)
            a += betas[(size_t)b * 11 + 1 + l] * JtJS[72 + tid * NB_ + l];
        Js[tid] = a * betas[(size_t)b * 11];
    }
    if (tid < NJ_) {                     // Rodrigues per joint
        int j = tid;
        float r0, r1, r2;
        if (j == 0) {
            r0 = gorient[(size_t)b * 3];
            r1 = gorient[(size_t)b * 3 + 1];
            r2 = gorient[(size_t)b * 3 + 2];
        } else {
            const float* p = body_pose + (size_t)b * 69 + (size_t)(j - 1) * 3;
            r0 = p[0]; r1 = p[1]; r2 = p[2];
        }
        float a0 = r0 + 1e-8f, a1 = r1 + 1e-8f, a2 = r2 + 1e-8f;
        float ang = sqrtf(a0 * a0 + a1 * a1 + a2 * a2);
        float inv = 1.f / ang;
        float ax = r0 * inv, ay = r1 * inv, az = r2 * inv;
        float cn = cosf(ang), sn = sinf(ang), omc = 1.f - cn;
        float K[9] = {0.f, -az, ay, az, 0.f, -ax, -ay, ax, 0.f};
        float K2[9];
#pragma unroll
        for (int i = 0; i < 3; ++i)
#pragma unroll
            for (int c = 0; c < 3; ++c) {
                float acc = 0.f;
#pragma unroll
                for (int m = 0; m < 3; ++m) acc += K[i * 3 + m] * K[m * 3 + c];
                K2[i * 3 + c] = acc;
            }
#pragma unroll
        for (int i = 0; i < 3; ++i)
#pragma unroll
            for (int c = 0; c < 3; ++c)
                Rm[j * 9 + i * 3 + c] =
                    (i == c ? 1.f : 0.f) + sn * K[i * 3 + c] + omc * K2[i * 3 + c];
    }
    __syncthreads();

    if (tid == 0) {                      // serial 24-joint chain (3x4 compose)
#pragma unroll
        for (int i = 0; i < 3; ++i) {
#pragma unroll
            for (int c = 0; c < 3; ++c) Ch[i * 4 + c] = Rm[i * 3 + c];
            Ch[i * 4 + 3] = Js[i];
        }
        for (int j = 1; j < NJ_; ++j) {
            int p = c_par[j];
            float rel[3];
#pragma unroll
            for (int c = 0; c < 3; ++c) rel[c] = Js[j * 3 + c] - Js[p * 3 + c];
#pragma unroll
            for (int i = 0; i < 3; ++i) {
#pragma unroll
                for (int c = 0; c < 3; ++c) {
                    float acc = 0.f;
#pragma unroll
                    for (int m = 0; m < 3; ++m)
                        acc += Ch[p * 12 + i * 4 + m] * Rm[j * 9 + m * 3 + c];
                    Ch[j * 12 + i * 4 + c] = acc;
                }
                float t = Ch[p * 12 + i * 4 + 3];
#pragma unroll
                for (int m = 0; m < 3; ++m) t += Ch[p * 12 + i * 4 + m] * rel[m];
                Ch[j * 12 + i * 4 + 3] = t;
            }
        }
    }
    __syncthreads();

    if (tid < 72) {                      // A = chain with t -= R @ J_shaped; posed joints out
        int j = tid / 3, row = tid % 3;
        const float* cr = &Ch[j * 12 + row * 4];
        float corr = cr[0] * Js[j * 3] + cr[1] * Js[j * 3 + 1] + cr[2] * Js[j * 3 + 2];
        float* A = A_ws + (size_t)b * 288 + j * 12 + row * 4;
        A[0] = cr[0]; A[1] = cr[1]; A[2] = cr[2]; A[3] = cr[3] - corr;
        out[((size_t)b * NOUT_ + V_ + j) * 3 + row] = cr[3] + transl[(size_t)b * 3 + row];
    }
}

// ---------------------------------------------------------------------------
// K4: fused LBS via split-f16 WMMA. Per 16-vertex tile:
//   T(16x12,f32) = Whi@Ahi + Wlo@Ahi + Whi@Alo   (3 x v_wmma_f32_16x16x32_f16)
// giving ~f32 accuracy. Each wave owns 4 tiles (64 verts); B-matrix (A-mats)
// f16 splits are computed once per wave and reused across tiles. Epilogue
// recomputes v_shaped in-register (shapedirs L2-resident) with all 32 lanes
// (2 vertices per lane), applies T + transl, writes vertices to the output.
// Block = 256 threads = 8 waves = 512 vertices; grid = (14, 512).
// ---------------------------------------------------------------------------
__global__ void k4_lbs(const float* __restrict__ betas,
                       const float* __restrict__ transl,
                       const float* __restrict__ scale,
                       const float* __restrict__ smpl,
                       const float* __restrict__ smil,
                       const float* __restrict__ SD,
                       const float* __restrict__ lbsW,
                       const float* __restrict__ A_ws,
                       float* __restrict__ out) {
    int b = blockIdx.y;
    int tid = threadIdx.x;
    int wave = tid >> 5, lane = tid & 31;

    __shared__ float sA[288];                    // A matrices (24 x 12, f32)
    __shared__ float sBeta[12];
    __shared__ float sT[8][TILES_ * 16 * 17];    // per-wave D spill, stride 17

    for (int i = tid; i < 288; i += 256) sA[i] = A_ws[(size_t)b * 288 + i];
    if (tid < 11) sBeta[tid] = betas[(size_t)b * 11 + tid];
    __syncthreads();

    int kb  = (lane >> 4) * 8;        // K-chunk base per ISA 16-bit A/B layout
    int col = lane & 15;
    bool cok = col < 12;

    // B-matrix (A-mats) split into hi/lo f16 halves — loop-invariant per wave.
    v16h b_hi, b_lo;
#pragma unroll
    for (int t = 0; t < 8; ++t) {
        float f = cok ? sA[(kb + t) * 12 + col] : 0.f;
        _Float16 h = (_Float16)f;
        b_hi[t] = h;
        b_lo[t] = (_Float16)(f - (float)h);
    }
#pragma unroll
    for (int t = 0; t < 8; ++t) {
        int kk = kb + 16 + t;                    // K pad 24 -> 32
        float f = (cok && kk < NJ_) ? sA[kk * 12 + col] : 0.f;
        _Float16 h = (_Float16)f;
        b_hi[8 + t] = h;
        b_lo[8 + t] = (_Float16)(f - (float)h);
    }

    int base0 = blockIdx.x * VPB_ + wave * (TILES_ * 16);
    int mb = (lane >> 4) * 8;         // D layout: lane<16 -> M=r, lane>=16 -> M=r+8

#pragma unroll
    for (int t = 0; t < TILES_; ++t) {
        int row = base0 + t * 16 + (lane & 15);
        int rc  = row < V_ ? row : V_ - 1;       // clamp: EXEC stays all-1 for WMMA
        const float* wr = lbsW + (size_t)rc * NJ_;

        v16h a_hi, a_lo;
#pragma unroll
        for (int q = 0; q < 8; ++q) {
            float f = wr[kb + q];
            _Float16 h = (_Float16)f;
            a_hi[q] = h;
            a_lo[q] = (_Float16)(f - (float)h);
        }
#pragma unroll
        for (int q = 0; q < 8; ++q) {
            int kk = kb + 16 + q;
            float f = (kk < NJ_) ? wr[kk] : 0.f;
            _Float16 h = (_Float16)f;
            a_hi[8 + q] = h;
            a_lo[8 + q] = (_Float16)(f - (float)h);
        }

        v8f acc = {};
        acc = __builtin_amdgcn_wmma_f32_16x16x32_f16(false, a_hi, false, b_hi,
                                                     (short)0, acc, false, false);
        acc = __builtin_amdgcn_wmma_f32_16x16x32_f16(false, a_lo, false, b_hi,
                                                     (short)0, acc, false, false);
        acc = __builtin_amdgcn_wmma_f32_16x16x32_f16(false, a_hi, false, b_lo,
                                                     (short)0, acc, false, false);

#pragma unroll
        for (int r = 0; r < 8; ++r)
            sT[wave][t * 272 + (mb + r) * 17 + col] = acc[r];
    }
    __syncthreads();

    float s  = scale[0];
    float b0 = sBeta[0];
#pragma unroll
    for (int t2 = 0; t2 < 2; ++t2) {             // 2 vertices per lane, all 32 lanes
        int vl = lane + 32 * t2;                 // 0..63 within wave's chunk
        int v  = base0 + vl;
        if (v < V_) {
            float vs[3];
#pragma unroll
            for (int k = 0; k < 3; ++k) {
                float vt = smpl[(size_t)v * 3 + k] * s + (1.f - s) * smil[(size_t)v * 3 + k];
                const float* sd = SD + ((size_t)v * 3 + k) * NB_;
                float bs = 0.f;
#pragma unroll
                for (int l = 0; l < NB_; ++l) bs += sBeta[1 + l] * sd[l];
                vs[k] = (vt + bs) * b0;          // v_shaped recomputed in-register
            }
            const float* T = &sT[wave][(vl >> 4) * 272 + (vl & 15) * 17];
#pragma unroll
            for (int i = 0; i < 3; ++i) {
                float o = T[i * 4] * vs[0] + T[i * 4 + 1] * vs[1] + T[i * 4 + 2] * vs[2]
                        + T[i * 4 + 3] + transl[(size_t)b * 3 + i];
                out[((size_t)b * NOUT_ + v) * 3 + i] = o;
            }
        }
    }
}

// ---------------------------------------------------------------------------
// K5: extra joints = J_regressor_extra @ verts. One block per batch; verts are
// read back from the output (translation passes through since softmax rows
// sum to 1). Writes the last 9 joint rows per batch.
// ---------------------------------------------------------------------------
__global__ void k5_extra(const float* __restrict__ JRE, float* __restrict__ out) {
    int b = blockIdx.x;
    int tid = threadIdx.x;               // 256
    float acc[27];
#pragma unroll
    for (int c = 0; c < 27; ++c) acc[c] = 0.f;
    const float* vb = out + (size_t)b * NOUT_ * 3;
    for (int v = tid; v < V_; v += 256) {
        float x = vb[(size_t)v * 3], y = vb[(size_t)v * 3 + 1], z = vb[(size_t)v * 3 + 2];
#pragma unroll
        for (int j = 0; j < NEX_; ++j) {
            float w = JRE[(size_t)j * V_ + v];
            acc[j * 3]     += w * x;
            acc[j * 3 + 1] += w * y;
            acc[j * 3 + 2] += w * z;
        }
    }
    __shared__ float red[256 * 27];
#pragma unroll
    for (int c = 0; c < 27; ++c) red[tid * 27 + c] = acc[c];
    __syncthreads();
    for (int off = 128; off > 0; off >>= 1) {
        if (tid < off) {
#pragma unroll
            for (int c = 0; c < 27; ++c)
                red[tid * 27 + c] += red[(tid + off) * 27 + c];
        }
        __syncthreads();
    }
    if (tid < 27) {
        int j = tid / 3, k = tid % 3;
        out[((size_t)b * NOUT_ + V_ + NJ_ + j) * 3 + k] = red[tid];
    }
}

// ---------------------------------------------------------------------------
extern "C" void kernel_launch(void* const* d_in, const int* in_sizes, int n_in,
                              void* d_out, int out_size, void* d_ws, size_t ws_size,
                              hipStream_t stream) {
    const float* betas     = (const float*)d_in[0];
    const float* body_pose = (const float*)d_in[1];
    const float* gorient   = (const float*)d_in[2];
    const float* transl    = (const float*)d_in[3];
    const float* scale     = (const float*)d_in[4];
    const float* smpl_t    = (const float*)d_in[5];
    const float* smil_t    = (const float*)d_in[6];
    const float* SD        = (const float*)d_in[7];
    const float* Jreg      = (const float*)d_in[8];
    const float* lbsW      = (const float*)d_in[9];
    const float* JRE       = (const float*)d_in[10];
    // d_in[11] = parents (hardcoded as __constant__ to match the reference)

    float* out  = (float*)d_out;
    float* ws   = (float*)d_ws;
    float* JtJS = ws;            // 792 floats
    float* A_ws = ws + 800;      // 512 * 288 floats (~590 KB)

    k1_jreg<<<72, 256, 0, stream>>>(scale, smpl_t, smil_t, SD, Jreg, JtJS);
    k3_chain<<<B_, 128, 0, stream>>>(betas, body_pose, gorient, transl, JtJS, A_ws, out);
    dim3 g4((V_ + VPB_ - 1) / VPB_, B_);
    k4_lbs<<<g4, 256, 0, stream>>>(betas, transl, scale, smpl_t, smil_t, SD, lbsW, A_ws, out);
    k5_extra<<<B_, 256, 0, stream>>>(JRE, out);
}
